// DecoderCrossAttention_57501022159095
// MI455X (gfx1250) — matrix operand order repaired and checked by reading, not compiled, not run on hardware
//
#include <hip/hip_runtime.h>

typedef __attribute__((ext_vector_type(16))) __bf16 v16bf;
typedef __attribute__((ext_vector_type(8)))  float  v8f;

// ---------------- workspace byte offsets ----------------
#define OFF_WQ   0u          // 512KB  W_q   B-frag [K512][N512]
#define OFF_WKG  524288u     // 512KB  W_kg  B-frag
#define OFF_WVG  1048576u    // 512KB  W_vg  B-frag
#define OFF_WOUT 1572864u    // 512KB  W_out B-frag
#define OFF_WKST 2097152u    // 512KB  per-head W_ks^T B-frag [d64][Kf512]
#define OFF_M    2621440u    // 4MB    per-head M[h]=Wvs_h*Wout_h  B-frag [512][512]
#define OFF_KG   6815744u    // 1MB    per (b,h) k_g B-frag [d64][l256]
#define OFF_VG   7864320u    // 1MB    v_g row-major bf16 [b][l][512]
#define OFF_G    8912896u    // 8MB    per (b,h) G=v_g_h*Wout_h B-frag [l256][o512]

__device__ __forceinline__ unsigned short f2bf(float f){
  unsigned u = __float_as_uint(f);
  u += 0x7FFFu + ((u >> 16) & 1u);
  return (unsigned short)(u >> 16);
}
__device__ __forceinline__ float bf2f(unsigned short s){
  return __uint_as_float(((unsigned)s) << 16);
}
__device__ __forceinline__ v8f wmma_bf16(v16bf a, v16bf b, v8f c){
  return __builtin_amdgcn_wmma_f32_16x16x32_bf16(false, a, false, b, (short)0, c, false, false);
}
// A-frag (16-bit 16x32) inverse maps: element (m, Klocal) -> (lane, j)
__device__ __forceinline__ int aLane(int m, int K){ return (m & 15) + (((K >> 4) & 1) << 4); }
__device__ __forceinline__ int aElem(int K){ return K & 15; }
// A-frag forward: (laneHi, j) -> Klocal
__device__ __forceinline__ int aKf(int hi, int j){ return j + (hi << 4); }

// =======================================================================
// K0: swizzle 5 weight views (f32 -> bf16 WMMA B-fragment layout) into ws
// =======================================================================
__global__ void __launch_bounds__(256, 1)
k0_swizzle(const float* __restrict__ Wq,  const float* __restrict__ Wks,
           const float* __restrict__ Wkg, const float* __restrict__ Wvg,
           const float* __restrict__ Wout, char* __restrict__ ws){
  int gid = blockIdx.x * 256 + threadIdx.x;      // 5 * 262144 total
  int mat = gid >> 18;
  int r   = gid & 262143;
  float val;
  unsigned short* dst;
  if (mat < 4){
    const float* W = (mat == 0) ? Wq : (mat == 1) ? Wkg : (mat == 2) ? Wvg : Wout;
    unsigned off   = (mat == 0) ? OFF_WQ : (mat == 1) ? OFF_WKG : (mat == 2) ? OFF_WVG : OFF_WOUT;
    int t = r >> 9, lane = (r >> 4) & 31, j = r & 15;
    int kt = t >> 5, nt = t & 31;
    int K = kt * 32 + (lane >> 4) * 16 + j;
    int n = nt * 16 + (lane & 15);
    val = W[K * 512 + n];
    dst = (unsigned short*)(ws + off) + r;
  } else { // W_ks^T per head: element (d, Kf) = Wks[Kf, h*64+d]
    int h = r >> 15, r2 = r & 32767;
    int t = r2 >> 9, lane = (r2 >> 4) & 31, j = r2 & 15;
    int kt = t >> 5, nt = t & 31;                // kt in 0..1
    int d  = kt * 32 + (lane >> 4) * 16 + j;
    int Kf = nt * 16 + (lane & 15);
    val = Wks[Kf * 512 + h * 64 + d];
    dst = (unsigned short*)(ws + OFF_WKST) + r;
  }
  *dst = f2bf(val);
}

// =======================================================================
// K0b: M[h] = Wvs[:, h*64:h*64+64] @ Wout[h*64:h*64+64, :]   (512x512/head)
// =======================================================================
__global__ void __launch_bounds__(256, 1)
k0b_build_M(const float* __restrict__ Wvs, char* __restrict__ ws){
  extern __shared__ char sm[];                   // 8KB A-frags (4mt x 2kt)
  const int tid = threadIdx.x, lane = tid & 31, w = tid >> 5;
  const int h = blockIdx.x >> 3, m0 = (blockIdx.x & 7) * 64;
  for (int e = tid; e < 4096; e += 256){
    int ml = e >> 6, dl = e & 63;
    float v = Wvs[(m0 + ml) * 512 + h * 64 + dl];
    ((unsigned short*)sm)[((ml >> 4) * 2 + (dl >> 5)) * 512 + aLane(ml, dl & 31) * 16 + aElem(dl & 31)] = f2bf(v);
  }
  __syncthreads();
  const v8f vzero = {0,0,0,0,0,0,0,0};
  v8f acc[4][4];
  #pragma unroll
  for (int a = 0; a < 4; a++)
    #pragma unroll
    for (int c = 0; c < 4; c++) acc[a][c] = vzero;
  #pragma unroll
  for (int kt = 0; kt < 2; kt++){
    v16bf af[4];
    #pragma unroll
    for (int mt = 0; mt < 4; mt++)
      af[mt] = *(const v16bf*)(sm + (((mt * 2 + kt) * 512 + lane * 16) << 1));
    #pragma unroll
    for (int i = 0; i < 4; i++){
      int nt = w * 4 + i;
      v16bf bb = *(const v16bf*)(ws + OFF_WOUT + (((((2 * h + kt) * 32 + nt) * 512) + lane * 16) << 1));
      #pragma unroll
      for (int mt = 0; mt < 4; mt++) acc[mt][i] = wmma_bf16(af[mt], bb, acc[mt][i]);
    }
  }
  unsigned short* M = (unsigned short*)(ws + OFF_M) + (unsigned)h * 262144u;
  #pragma unroll
  for (int mt = 0; mt < 4; mt++)
    #pragma unroll
    for (int i = 0; i < 4; i++){
      int nt = w * 4 + i;
      int o  = nt * 16 + (lane & 15);
      #pragma unroll
      for (int rr = 0; rr < 8; rr++){
        int Kf = m0 + mt * 16 + rr + 8 * (lane >> 4);
        M[((Kf >> 5) * 32 + nt) * 512 + ((o & 15) + 16 * ((Kf >> 4) & 1)) * 16 + (Kf & 15)] = f2bf(acc[mt][i][rr]);
      }
    }
}

// =======================================================================
// K1: k_g = gc@W_kg (-> per (b,h) score B-frags), v_g = gc@W_vg (row-major bf16)
// =======================================================================
__global__ void __launch_bounds__(256, 1)
k1_kgvg(const float* __restrict__ gc, char* __restrict__ ws){
  extern __shared__ char sm[];                   // 64KB A-frags (4mt x 16kt)
  const int tid = threadIdx.x, lane = tid & 31, w = tid >> 5;
  const int b = blockIdx.x >> 2, l0 = (blockIdx.x & 3) * 64;
  for (int p = tid; p < 16384; p += 256){
    int ml = p >> 8, kp = (p & 255) * 2;
    const float* src = gc + ((b * 256 + l0 + ml) * 512 + kp);
    unsigned pk = (unsigned)f2bf(src[0]) | ((unsigned)f2bf(src[1]) << 16);
    int mt = ml >> 4, kt = kp >> 5, Kl = kp & 31;
    *(unsigned*)(sm + (((mt * 16 + kt) * 512 + aLane(ml, Kl) * 16 + aElem(Kl)) << 1)) = pk;
  }
  __syncthreads();
  const v8f vzero = {0,0,0,0,0,0,0,0};
  for (int which = 0; which < 2; ++which){
    unsigned woff = which ? OFF_WVG : OFF_WKG;
    v8f acc[4][4];
    #pragma unroll
    for (int a = 0; a < 4; a++)
      #pragma unroll
      for (int c = 0; c < 4; c++) acc[a][c] = vzero;
    for (int kt = 0; kt < 16; ++kt){
      v16bf af[4];
      #pragma unroll
      for (int mt = 0; mt < 4; mt++)
        af[mt] = *(const v16bf*)(sm + (((mt * 16 + kt) * 512 + lane * 16) << 1));
      #pragma unroll
      for (int i = 0; i < 4; i++){
        int nt = w * 4 + i;
        v16bf bb = *(const v16bf*)(ws + woff + ((((kt * 32 + nt) * 512) + lane * 16) << 1));
        #pragma unroll
        for (int mt = 0; mt < 4; mt++) acc[mt][i] = wmma_bf16(af[mt], bb, acc[mt][i]);
      }
    }
    #pragma unroll
    for (int mt = 0; mt < 4; mt++)
      #pragma unroll
      for (int i = 0; i < 4; i++){
        int f = (w * 4 + i) * 16 + (lane & 15);
        #pragma unroll
        for (int rr = 0; rr < 8; rr++){
          int lg = l0 + mt * 16 + rr + 8 * (lane >> 4);
          unsigned short v = f2bf(acc[mt][i][rr]);
          if (which == 0){
            int h = f >> 6, d = f & 63;
            unsigned short* kg = (unsigned short*)(ws + OFF_KG) + (unsigned)(b * 8 + h) * 16384u;
            kg[((d >> 5) * 16 + (lg >> 4)) * 512 + ((lg & 15) + 16 * ((d >> 4) & 1)) * 16 + (d & 15)] = v;
          } else {
            ((unsigned short*)(ws + OFF_VG))[(unsigned)(b * 256 + lg) * 512u + f] = v;
          }
        }
      }
  }
}

// =======================================================================
// K1b: G[b,h] = v_g[b,:,h,:] @ Wout[h*64:h*64+64,:]   (256x512 per b,h)
// =======================================================================
__global__ void __launch_bounds__(256, 1)
k1b_build_G(char* __restrict__ ws){
  extern __shared__ char sm[];                   // 8KB A-frags
  const int tid = threadIdx.x, lane = tid & 31, w = tid >> 5;
  const int b = blockIdx.x >> 5, h = (blockIdx.x >> 2) & 7, l0 = (blockIdx.x & 3) * 64;
  const unsigned short* vg = (const unsigned short*)(ws + OFF_VG);
  for (int e = tid; e < 4096; e += 256){
    int ml = e >> 6, dl = e & 63;
    unsigned short v = vg[(unsigned)(b * 256 + l0 + ml) * 512u + h * 64 + dl];
    ((unsigned short*)sm)[((ml >> 4) * 2 + (dl >> 5)) * 512 + aLane(ml, dl & 31) * 16 + aElem(dl & 31)] = v;
  }
  __syncthreads();
  const v8f vzero = {0,0,0,0,0,0,0,0};
  v8f acc[4][4];
  #pragma unroll
  for (int a = 0; a < 4; a++)
    #pragma unroll
    for (int c = 0; c < 4; c++) acc[a][c] = vzero;
  #pragma unroll
  for (int kt = 0; kt < 2; kt++){
    v16bf af[4];
    #pragma unroll
    for (int mt = 0; mt < 4; mt++)
      af[mt] = *(const v16bf*)(sm + (((mt * 2 + kt) * 512 + lane * 16) << 1));
    #pragma unroll
    for (int i = 0; i < 4; i++){
      int nt = w * 4 + i;
      v16bf bb = *(const v16bf*)(ws + OFF_WOUT + (((((2 * h + kt) * 32 + nt) * 512) + lane * 16) << 1));
      #pragma unroll
      for (int mt = 0; mt < 4; mt++) acc[mt][i] = wmma_bf16(af[mt], bb, acc[mt][i]);
    }
  }
  unsigned short* G = (unsigned short*)(ws + OFF_G) + (unsigned)(b * 8 + h) * 131072u;
  #pragma unroll
  for (int mt = 0; mt < 4; mt++)
    #pragma unroll
    for (int i = 0; i < 4; i++){
      int nt = w * 4 + i;
      int o  = nt * 16 + (lane & 15);
      #pragma unroll
      for (int rr = 0; rr < 8; rr++){
        int lg = l0 + mt * 16 + rr + 8 * (lane >> 4);
        G[((lg >> 5) * 32 + nt) * 512 + ((o & 15) + 16 * ((lg >> 4) & 1)) * 16 + (lg & 15)] = f2bf(acc[mt][i][rr]);
      }
    }
}

// =======================================================================
// K2: fused attention per 8-token workgroup. grid: b(4) x 128 groups.
// Double-buffered ctx staging (SCA0/SCB0 vs SCA1/SCB1).
// =======================================================================
#define SQ    0        // 16KB  q A-frags per head [h][kt2] (M=8 padded)
#define SQWK  16384    // 64KB  qWk row-major bf16 [tok8][h8][512]
#define SATT  81920    // 32KB  scores_g -> attn_g bf16 [tok8][h8][256]
#define SACT  114688   // 64KB  actx row-major bf16 [tok8][h8][512]
#define SCA0  180224   // 32KB  ctx A-frags buf0 [2mt x 16kt]  (stage A: x frags)
#define SCB0  212992   // 32KB  ctx B-frags buf0 [32nt]
#define SCA1  245760   // 32KB  ctx A-frags buf1
#define SCB1  278528   // 32KB  ctx B-frags buf1
#define SSF   311296   // 1KB   attn_s A-frag (pad lanes zeroed once)
#define SSS   312320   // 1KB   spatial scores f32 [h8][32]
#define SMEM_K2 313344

// stream one token's ctx (32x512 f32) from HBM -> bf16 A-frags + B-frags in LDS.
// Also prefetch (global_prefetch_b8) the ctx eight tokens ahead into L2.
__device__ __forceinline__ void load_ctx(char* sm, const float* __restrict__ sc,
                                         size_t base, int buf, int t0, int stride){
  char* ca = sm + (buf ? SCA1 : SCA0);
  unsigned short* cb = (unsigned short*)(sm + (buf ? SCB1 : SCB0));
  for (int p = t0; p < 8192; p += stride){
    int k = p >> 8, kp = (p & 255) * 2;
    const float* src = sc + base + (size_t)k * 512 + kp;
    unsigned short b0 = f2bf(src[0]), b1 = f2bf(src[1]);
    int mt = k >> 4, kt = kp >> 5, Kl = kp & 31;
    *(unsigned*)(ca + ((((mt * 16 + kt) * 512) + aLane(k, Kl) * 16 + aElem(Kl)) << 1))
        = (unsigned)b0 | ((unsigned)b1 << 16);
    int f0 = kp, f1 = kp + 1;
    cb[(f0 >> 4) * 512 + ((f0 & 15) + 16 * (k >> 4)) * 16 + (k & 15)] = b0;
    cb[(f1 >> 4) * 512 + ((f1 & 15) + 16 * (k >> 4)) * 16 + (k & 15)] = b1;
    if ((p & 15) == 0){
      size_t pfo = base + (size_t)k * 512 + kp + 131072; // 8 tokens ahead
      if (pfo < (size_t)67108864)
        __builtin_prefetch((const void*)(sc + pfo), 0, 3);
    }
  }
}

__global__ void __launch_bounds__(256, 1)
k2_fused(const float* __restrict__ x, const float* __restrict__ sc,
         const float* __restrict__ bout, const char* __restrict__ ws,
         float* __restrict__ out){
  extern __shared__ char sm[];
  const int tid = threadIdx.x, lane = tid & 31, w = tid >> 5;
  const int b = blockIdx.x >> 7, n0 = (blockIdx.x & 127) * 8;
  const v8f vzero = {0,0,0,0,0,0,0,0};

  // ---- A0/A1: zero pad regions (SQ, x-frags, SSF once), load x -> A-frags
  for (int i = tid; i < 4096; i += 256) ((unsigned*)(sm + SQ))[i] = 0;
  for (int i = tid; i < 4096; i += 256) ((unsigned*)(sm + SCA0))[i] = 0;
  ((unsigned*)(sm + SSF))[tid] = 0;
  __syncthreads();
  for (int p = tid; p < 2048; p += 256){
    int tok = p >> 8, kp = (p & 255) * 2;
    const float* src = x + ((b * 1024 + n0 + tok) * 512 + kp);
    unsigned pk = (unsigned)f2bf(src[0]) | ((unsigned)f2bf(src[1]) << 16);
    int kt = kp >> 5, Kl = kp & 31;
    *(unsigned*)(sm + SCA0 + ((kt * 512 + aLane(tok, Kl) * 16 + aElem(Kl)) << 1)) = pk;
  }
  __syncthreads();

  // ---- A2: q = 0.125 * x @ W_q  ->  per-head A-frags in SQ
  {
    v8f acc[4];
    #pragma unroll
    for (int i = 0; i < 4; i++) acc[i] = vzero;
    for (int kt = 0; kt < 16; kt++){
      v16bf a = *(const v16bf*)(sm + SCA0 + ((kt * 512 + lane * 16) << 1));
      #pragma unroll
      for (int i = 0; i < 4; i++){
        int nt = w * 4 + i;
        v16bf bb = *(const v16bf*)(ws + OFF_WQ + ((((kt * 32 + nt) * 512) + lane * 16) << 1));
        acc[i] = wmma_bf16(a, bb, acc[i]);
      }
    }
    #pragma unroll
    for (int i = 0; i < 4; i++)
      #pragma unroll
      for (int rr = 0; rr < 8; rr++){
        int m = rr + 8 * (lane >> 4);
        if (m < 8){
          int f = (w * 4 + i) * 16 + (lane & 15);
          int h = f >> 6, dd = f & 63;
          ((unsigned short*)(sm + SQ))[(h * 2 + (dd >> 5)) * 512 + aLane(m, dd & 31) * 16 + aElem(dd & 31)]
              = f2bf(acc[i][rr] * 0.125f);
        }
      }
  }
  __syncthreads();

  // ---- A3: scores_g per head (wave == head) -> SATT bf16
  {
    const int h = w;
    for (int nt = 0; nt < 16; nt++){
      v8f acc = vzero;
      #pragma unroll
      for (int kt = 0; kt < 2; kt++){
        v16bf a  = *(const v16bf*)(sm + SQ + (((h * 2 + kt) * 512 + lane * 16) << 1));
        v16bf bb = *(const v16bf*)(ws + OFF_KG +
                     ((unsigned)(b * 8 + h) * 32768u) + ((((kt * 16 + nt) * 512) + lane * 16) << 1));
        acc = wmma_bf16(a, bb, acc);
      }
      #pragma unroll
      for (int rr = 0; rr < 8; rr++){
        int m = rr + 8 * (lane >> 4);
        if (m < 8){
          int l = nt * 16 + (lane & 15);
          ((unsigned short*)(sm + SATT))[(m * 8 + h) * 256 + l] = f2bf(acc[rr]);
        }
      }
    }
  }
  // ---- A4: qWk per head (wave == head) -> SQWK row-major bf16
  {
    const int h = w;
    for (int nt = 0; nt < 32; nt++){
      v8f acc = vzero;
      #pragma unroll
      for (int kt = 0; kt < 2; kt++){
        v16bf a  = *(const v16bf*)(sm + SQ + (((h * 2 + kt) * 512 + lane * 16) << 1));
        v16bf bb = *(const v16bf*)(ws + OFF_WKST +
                     ((unsigned)h * 65536u) + ((((kt * 32 + nt) * 512) + lane * 16) << 1));
        acc = wmma_bf16(a, bb, acc);
      }
      #pragma unroll
      for (int rr = 0; rr < 8; rr++){
        int m = rr + 8 * (lane >> 4);
        if (m < 8){
          int Kf = nt * 16 + (lane & 15);
          ((unsigned short*)(sm + SQWK))[(m * 8 + h) * 512 + Kf] = f2bf(acc[rr]);
        }
      }
    }
  }
  __syncthreads();

  // ---- prologue: stage token 0 into buffer 0 (all threads)
  const size_t ctxTok = 16384; // 32*512 floats per token
  load_ctx(sm, sc, (size_t)(b * 1024 + n0) * ctxTok / 1, 0, tid, 256); // base in floats
  __syncthreads();

  // ================= per-token passes (double buffered) =================
  #pragma unroll 1
  for (int t = 0; t < 8; t++){
    const int buf = t & 1;
    char* ca = sm + (buf ? SCA1 : SCA0);
    char* cbb = sm + (buf ? SCB1 : SCB0);
    // P2: scores_s = ctx @ qWk^T on waves 0,1; waves 2..7 stream token t+1
    if (w < 2){
      const int mt = w;
      v8f acc = vzero;
      for (int kt = 0; kt < 16; kt++){
        v16bf a = *(const v16bf*)(ca + ((((mt * 16 + kt) * 512) + lane * 16) << 1));
        union { v16bf v; unsigned short u[16]; } bbv;
        int hh = lane & 15, hiK = lane >> 4;
        const unsigned short* qwk = (const unsigned short*)(sm + SQWK) + (t * 8 + (hh & 7)) * 512;
        #pragma unroll
        for (int j = 0; j < 16; j++)
          bbv.u[j] = (hh < 8) ? qwk[kt * 32 + hiK * 16 + j] : (unsigned short)0;
        acc = wmma_bf16(a, bbv.v, acc);
      }
      #pragma unroll
      for (int rr = 0; rr < 8; rr++){
        int hh = lane & 15;
        if (hh < 8){
          int k = mt * 16 + rr + 8 * (lane >> 4);
          ((float*)(sm + SSS))[hh * 32 + k] = acc[rr];
        }
      }
    } else if (t < 7){
      load_ctx(sm, sc, (size_t)(b * 1024 + n0 + t + 1) * ctxTok, buf ^ 1, tid - 64, 192);
    }
    __syncthreads();
    // P3: joint softmax over [32 spatial | 256 global] (wave == head)
    {
      const int h = w;
      const float* ss = (const float*)(sm + SSS) + h * 32;
      unsigned short* sg = (unsigned short*)(sm + SATT) + (t * 8 + h) * 256;
      float vsp = ss[lane];
      float mx = vsp, g[8];
      #pragma unroll
      for (int i = 0; i < 8; i++){ g[i] = bf2f(sg[lane + 32 * i]); mx = fmaxf(mx, g[i]); }
      #pragma unroll
      for (int o = 16; o >= 1; o >>= 1) mx = fmaxf(mx, __shfl_xor(mx, o, 32));
      float es = __expf(vsp - mx), sum = es, eg[8];
      #pragma unroll
      for (int i = 0; i < 8; i++){ eg[i] = __expf(g[i] - mx); sum += eg[i]; }
      #pragma unroll
      for (int o = 16; o >= 1; o >>= 1) sum += __shfl_xor(sum, o, 32);
      float inv = 1.0f / sum;
      int k = lane;  // attn_s -> A-frag (M=h, K=k); pad slots stay 0 from A0
      ((unsigned short*)(sm + SSF))[(h + 16 * ((k >> 4) & 1)) * 16 + aElem(k)] = f2bf(es * inv);
      #pragma unroll
      for (int i = 0; i < 8; i++) sg[lane + 32 * i] = f2bf(eg[i] * inv);
    }
    __syncthreads();
    // P4: actx = attn_s @ ctx  (all waves; M = heads padded)
    {
      v16bf a = *(const v16bf*)(sm + SSF + lane * 32);
      #pragma unroll
      for (int i = 0; i < 4; i++){
        int nt = w * 4 + i;
        v16bf bb = *(const v16bf*)(cbb + ((nt * 512 + lane * 16) << 1));
        v8f acc = wmma_bf16(a, bb, vzero);
        #pragma unroll
        for (int rr = 0; rr < 8; rr++){
          if (lane < 16){
            int f = nt * 16 + lane;
            ((unsigned short*)(sm + SACT))[(t * 8 + rr) * 512 + f] = f2bf(acc[rr]);
          }
        }
      }
    }
    __syncthreads();
  }

  // ====== folded output GEMM: out = sum_h attn_g@G[b,h] + sum_h actx@M[h] + b_out
  {
    v8f acc[4];
    #pragma unroll
    for (int i = 0; i < 4; i++) acc[i] = vzero;
    #pragma unroll 1
    for (int h = 0; h < 8; h++){
      for (int kt = 0; kt < 8; kt++){            // attn_g part (K=256)
        union { v16bf v; unsigned short u[16]; } a;
        int m = lane & 15, hi = lane >> 4;
        const unsigned short* at = (const unsigned short*)(sm + SATT) + ((m & 7) * 8 + h) * 256;
        #pragma unroll
        for (int j = 0; j < 16; j++)
          a.u[j] = (m < 8) ? at[kt * 32 + aKf(hi, j)] : (unsigned short)0;
        #pragma unroll
        for (int i = 0; i < 4; i++){
          int nt = w * 4 + i;
          v16bf bb = *(const v16bf*)(ws + OFF_G + ((unsigned)(b * 8 + h) * 262144u)
                       + ((((kt * 32 + nt) * 512) + lane * 16) << 1));
          acc[i] = wmma_bf16(a.v, bb, acc[i]);
        }
      }
      for (int kt = 0; kt < 16; kt++){           // actx part (K=512)
        union { v16bf v; unsigned short u[16]; } a;
        int m = lane & 15, hi = lane >> 4;
        const unsigned short* ax = (const unsigned short*)(sm + SACT) + ((m & 7) * 8 + h) * 512;
        #pragma unroll
        for (int j = 0; j < 16; j++)
          a.u[j] = (m < 8) ? ax[kt * 32 + aKf(hi, j)] : (unsigned short)0;
        #pragma unroll
        for (int i = 0; i < 4; i++){
          int nt = w * 4 + i;
          v16bf bb = *(const v16bf*)(ws + OFF_M + ((unsigned)h * 524288u)
                       + ((((kt * 32 + nt) * 512) + lane * 16) << 1));
          acc[i] = wmma_bf16(a.v, bb, acc[i]);
        }
      }
    }
    #pragma unroll
    for (int i = 0; i < 4; i++){
      int nt = w * 4 + i;
      int o  = nt * 16 + (lane & 15);
      float bo = bout[o];
      #pragma unroll
      for (int rr = 0; rr < 8; rr++){
        int m = rr + 8 * (lane >> 4);
        if (m < 8)
          out[(b * 1024 + n0 + m) * 512 + o] = acc[i][rr] + bo;
      }
    }
  }
}

extern "C" void kernel_launch(void* const* d_in, const int* in_sizes, int n_in,
                              void* d_out, int out_size, void* d_ws, size_t ws_size,
                              hipStream_t stream){
  const float* x    = (const float*)d_in[0];
  const float* sc   = (const float*)d_in[1];
  const float* gc   = (const float*)d_in[2];
  const float* Wq   = (const float*)d_in[3];
  const float* Wks  = (const float*)d_in[4];
  const float* Wvs  = (const float*)d_in[5];
  const float* Wkg  = (const float*)d_in[6];
  const float* Wvg  = (const float*)d_in[7];
  const float* Wout = (const float*)d_in[8];
  const float* bo   = (const float*)d_in[9];
  char* ws = (char*)d_ws;
  float* out = (float*)d_out;

  k0_swizzle <<<5120, 256, 0,       stream>>>(Wq, Wks, Wkg, Wvg, Wout, ws);
  k0b_build_M<<<64,   256, 8192,    stream>>>(Wvs, ws);
  k1_kgvg    <<<16,   256, 65536,   stream>>>(gc, ws);
  k1b_build_G<<<128,  256, 8192,    stream>>>(ws);
  k2_fused   <<<512,  256, SMEM_K2, stream>>>(x, sc, bo, ws, out);
}
